// DecoderLayer_57526791962890
// MI455X (gfx1250) — matrix operand order repaired and checked by reading, not compiled, not run on hardware
//
#include <hip/hip_runtime.h>
#include <hip/hip_bf16.h>

// ---------------------------------------------------------------------------
// CDNA5 (gfx1250) decoder layer: all matmuls via v_wmma_f32_16x16x32_f16.
// wave32 everywhere; fragment layouts follow CDNA5 ISA §7.12.2.
// GEMM uses 64x64 per-wave tiles (4x4 WMMA accums) to double WMMA:LDS ratio.
// ---------------------------------------------------------------------------

typedef __attribute__((ext_vector_type(16))) _Float16 v16h;
typedef __attribute__((ext_vector_type(8)))  _Float16 v8h;
typedef __attribute__((ext_vector_type(8)))  float    v8f;

static __device__ __forceinline__ v8f wmma_f16(v16h a, v16h b, v8f c) {
  return __builtin_amdgcn_wmma_f32_16x16x32_f16(
      /*neg_a=*/false, a, /*neg_b=*/false, b,
      /*c_mod=*/(short)0, c, /*reuse_a=*/false, /*reuse_b=*/false);
}

// A fragment (16x32 f16, MxK). Lane L<16 holds row M=L, K {0..7,16..23};
// lane L>=16 holds row M=L-16, K {8..15,24..31}.  base points at row 0, K=0.
static __device__ __forceinline__ v16h frag_a(const _Float16* base, int stride) {
  int lane = threadIdx.x & 31;
  int r = lane & 15, hf = lane >> 4;
  const _Float16* p = base + (long)r * stride;
  v8h lo = *(const v8h*)(p + hf * 8);
  v8h hi = *(const v8h*)(p + 16 + hf * 8);
  v16h a;
#pragma unroll
  for (int i = 0; i < 8; ++i) { a[i] = lo[i]; a[i + 8] = hi[i]; }
  return a;
}

// B fragment (32x16 f16, KxN), sourced from a buffer indexed [n][k] (k
// contiguous).  Lane L<16 holds column N=L, K {0..7,16..23}; lane L>=16 holds
// column N=L-16, K {8..15,24..31}.
static __device__ __forceinline__ v16h frag_b(const _Float16* base, int stride) {
  int lane = threadIdx.x & 31;
  int n = lane & 15, hf = lane >> 4;
  const _Float16* p = base + (long)n * stride;
  v8h lo = *(const v8h*)(p + hf * 8);
  v8h hi = *(const v8h*)(p + 16 + hf * 8);
  v16h b;
#pragma unroll
  for (int i = 0; i < 8; ++i) { b[i] = lo[i]; b[i + 8] = hi[i]; }
  return b;
}

// ---------------------------------------------------------------------------
// GEMM: C[M,N] = A[M,K](f16) @ W[N,K](f16)^T (+ res f32) -> out32 / out16.
// Block tile 256x128 (8 waves, 4 in M x 2 in N); each wave owns 64x64
// (4x4 WMMA accums -> 16 v_wmma per 16 ds_load_b128 per K-step).
// ---------------------------------------------------------------------------
__global__ __launch_bounds__(256) void gemm_wmma_kernel(
    const _Float16* __restrict__ A, const _Float16* __restrict__ W,
    const float* __restrict__ res, float* __restrict__ out32,
    _Float16* __restrict__ out16, int M, int N, int K) {
  __shared__ _Float16 As[256][40];  // 32 K + 8 pad (row = 80B, 16B aligned)
  __shared__ _Float16 Bs[128][40];
  int tid = threadIdx.x;
  int lane = tid & 31, wave = tid >> 5;
  int wm = wave & 3, wn = wave >> 2;   // 4 waves in M, 2 in N
  long mBase = (long)blockIdx.y * 256;
  long nBase = (long)blockIdx.x * 128;
  v8f acc[4][4] = {};

  for (int kk = 0; kk < K; kk += 32) {
    __syncthreads();
#pragma unroll
    for (int j = 0; j < 4; ++j) {  // A tile: thread tid loads row tid (64B)
      int c = tid * 4 + j;
      int r = c >> 2, k8 = (c & 3) << 3;
      *(v8h*)&As[r][k8] = *(const v8h*)(A + (mBase + r) * (long)K + kk + k8);
    }
#pragma unroll
    for (int j = 0; j < 2; ++j) {  // B tile: 512 chunks / 256 threads
      int c = tid * 2 + j;
      int r = c >> 2, k8 = (c & 3) << 3;
      *(v8h*)&Bs[r][k8] = *(const v8h*)(W + (nBase + r) * (long)K + kk + k8);
    }
    __syncthreads();
    v16h af[4], bf[4];
#pragma unroll
    for (int t = 0; t < 4; ++t) af[t] = frag_a(&As[wm * 64 + t * 16][0], 40);
#pragma unroll
    for (int t = 0; t < 4; ++t) bf[t] = frag_b(&Bs[wn * 64 + t * 16][0], 40);
#pragma unroll
    for (int i = 0; i < 4; ++i)
#pragma unroll
      for (int j = 0; j < 4; ++j)
        acc[i][j] = wmma_f16(af[i], bf[j], acc[i][j]);
  }

  int col0 = lane & 15, rh = (lane >> 4) << 3;
#pragma unroll
  for (int tm = 0; tm < 4; ++tm)
#pragma unroll
    for (int tn = 0; tn < 4; ++tn) {
      v8f c = acc[tm][tn];
#pragma unroll
      for (int i = 0; i < 8; ++i) {
        long row = mBase + wm * 64 + tm * 16 + i + rh;
        long col = nBase + wn * 64 + tn * 16 + col0;
        float v = c[i];
        if (res)   v += res[row * N + col];
        if (out32) out32[row * N + col] = v;
        if (out16) out16[row * N + col] = (_Float16)v;
      }
    }
}

// ---------------------------------------------------------------------------
// Flash attention (no 1/sqrt(d) scale, per reference).  Q/K/V: [B*H][S][64]
// f16.  Block = 4 waves, 64 q rows per (b,h); key blocks of 32 staged in LDS.
// Output written as [B][SQ][H*64] f16 (feeds output-projection GEMM).
// ---------------------------------------------------------------------------
template <bool CAUSAL>
__global__ __launch_bounds__(128, 2) void attn_wmma_kernel(
    const _Float16* __restrict__ Q, const _Float16* __restrict__ Km,
    const _Float16* __restrict__ Vm, _Float16* __restrict__ out,
    int SQv, int SKv, int Hn) {
  __shared__ _Float16 Ks[32][72];      // [key][d]   (row = 144B)
  __shared__ _Float16 Vt[64][40];      // [d][key]   (row = 80B)
  __shared__ _Float16 Ps[4][16][40];   // per-wave P staging, [row][key]
  int tid = threadIdx.x, lane = tid & 31, wave = tid >> 5;
  int bh = blockIdx.y;
  int b = bh / Hn, h = bh % Hn;
  const _Float16* Qp = Q + (long)bh * SQv * 64;
  const _Float16* Kp = Km + (long)bh * SKv * 64;
  const _Float16* Vp = Vm + (long)bh * SKv * 64;
  int q0 = blockIdx.x * 64;
  int qw = q0 + wave * 16;

  // Persistent Q fragments (d 0..31 and 32..63), straight from global.
  v16h aq0 = frag_a(Qp + (long)qw * 64, 64);
  v16h aq1 = frag_a(Qp + (long)qw * 64 + 32, 64);

  float m[8], l[8];
#pragma unroll
  for (int i = 0; i < 8; ++i) { m[i] = -1e30f; l[i] = 0.f; }
  v8f o0 = {}, o1 = {}, o2 = {}, o3 = {};

  int kmax = CAUSAL ? (q0 + 64) : SKv;
  if (kmax > SKv) kmax = SKv;
  int col = lane & 15, rh = (lane >> 4) << 3;

  for (int kb = 0; kb < kmax; kb += 32) {
    __syncthreads();
#pragma unroll
    for (int j = 0; j < 2; ++j) {  // K tile: 256 8-half chunks / 128 threads
      int c = tid * 2 + j;
      int key = c >> 3, d8 = (c & 7) << 3;
      *(v8h*)&Ks[key][d8] = *(const v8h*)(Kp + (long)(kb + key) * 64 + d8);
    }
#pragma unroll
    for (int j = 0; j < 16; ++j) {  // V tile, transposed into LDS
      int e = tid + 128 * j;
      int key = e >> 6, d = e & 63;
      Vt[d][key] = Vp[(long)(kb + key) * 64 + d];
    }
    __syncthreads();

    // S = Q . K^T  (two 16x16 score tiles: keys kb..+15 and kb+16..+31)
    v8f z = {};
    v8f s0 = wmma_f16(aq0, frag_b(&Ks[0][0], 72), z);
    s0 = wmma_f16(aq1, frag_b(&Ks[0][32], 72), s0);
    v8f s1 = wmma_f16(aq0, frag_b(&Ks[16][0], 72), z);
    s1 = wmma_f16(aq1, frag_b(&Ks[16][32], 72), s1);

    // Online softmax per row.  C-layout: element i of half (lane>>4) is row
    // i + 8*(lane>>4); the 16 lanes of a half hold one row's 16 columns, so
    // xor-shuffles with mask<16 reduce exactly within the row.
#pragma unroll
    for (int i = 0; i < 8; ++i) {
      int row = qw + i + rh;
      float x0 = s0[i], x1 = s1[i];
      if (CAUSAL) {
        if (kb + col > row)      x0 = -1e30f;
        if (kb + 16 + col > row) x1 = -1e30f;
      }
      float t = fmaxf(x0, x1);
#pragma unroll
      for (int msk = 8; msk; msk >>= 1) t = fmaxf(t, __shfl_xor(t, msk, 32));
      float mn = fmaxf(m[i], t);
      float sc = __expf(m[i] - mn);
      float p0 = __expf(x0 - mn), p1 = __expf(x1 - mn);
      float ps = p0 + p1;
#pragma unroll
      for (int msk = 8; msk; msk >>= 1) ps += __shfl_xor(ps, msk, 32);
      l[i] = l[i] * sc + ps;
      m[i] = mn;
      o0[i] *= sc; o1[i] *= sc; o2[i] *= sc; o3[i] *= sc;
      Ps[wave][i + rh][col]      = (_Float16)p0;  // C-layout -> A-layout via LDS
      Ps[wave][i + rh][16 + col] = (_Float16)p1;
    }

    // O += P . V   (K dim = 32 keys, N = 4 chunks of 16 d-columns)
    v16h ap = frag_a(&Ps[wave][0][0], 40);
    o0 = wmma_f16(ap, frag_b(&Vt[0][0], 40), o0);
    o1 = wmma_f16(ap, frag_b(&Vt[16][0], 40), o1);
    o2 = wmma_f16(ap, frag_b(&Vt[32][0], 40), o2);
    o3 = wmma_f16(ap, frag_b(&Vt[48][0], 40), o3);
  }

#pragma unroll
  for (int i = 0; i < 8; ++i) {
    float inv = 1.0f / l[i];
    long q = qw + i + rh;
    _Float16* orow = out + ((long)b * SQv + q) * (Hn * 64) + h * 64;
    orow[col]      = (_Float16)(o0[i] * inv);
    orow[16 + col] = (_Float16)(o1[i] * inv);
    orow[32 + col] = (_Float16)(o2[i] * inv);
    orow[48 + col] = (_Float16)(o3[i] * inv);
  }
}

// ---------------------------------------------------------------------------
// Elementwise / reshape helpers
// ---------------------------------------------------------------------------
__global__ __launch_bounds__(256) void rmsnorm_f16_kernel(
    const float* __restrict__ x, const float* __restrict__ scale,
    _Float16* __restrict__ out, int E) {
  __shared__ float red[256];
  int row = blockIdx.x, tid = threadIdx.x;
  const float* xr = x + (long)row * E;
  float s = 0.f;
  for (int e = tid; e < E; e += 256) { float v = xr[e]; s += v * v; }
  red[tid] = s;
  __syncthreads();
  for (int off = 128; off; off >>= 1) {
    if (tid < off) red[tid] += red[tid + off];
    __syncthreads();
  }
  float rs = rsqrtf(red[0] / (float)E + 1e-6f);
  for (int e = tid; e < E; e += 256)
    out[(long)row * E + e] = (_Float16)(xr[e] * rs * scale[e]);
}

__global__ __launch_bounds__(256) void f32_to_f16_kernel(
    const float* __restrict__ s, _Float16* __restrict__ d, size_t n) {
  size_t i = (size_t)blockIdx.x * 1024 + threadIdx.x;
#pragma unroll
  for (int j = 0; j < 4; ++j) {
    size_t k = i + (size_t)j * 256;
    if (k < n) d[k] = (_Float16)s[k];
  }
}

__global__ __launch_bounds__(256) void gelu_gate_kernel(
    const _Float16* __restrict__ h0, const _Float16* __restrict__ h1,
    _Float16* __restrict__ hg, size_t n) {
  size_t i = (size_t)blockIdx.x * 1024 + threadIdx.x;
#pragma unroll
  for (int j = 0; j < 4; ++j) {
    size_t k = i + (size_t)j * 256;
    if (k < n) {
      float a = (float)h0[k], g = (float)h1[k];
      float e = 0.5f * a * (1.0f + erff(a * 0.70710678118f));  // exact gelu
      hg[k] = (_Float16)(e * g);
    }
  }
}

// qkv [B*S][3*H*64] -> Q/K/V [B][H][S][64]
__global__ __launch_bounds__(256) void pack_qkv_kernel(
    const _Float16* __restrict__ qkv, _Float16* __restrict__ Qh,
    _Float16* __restrict__ Kh, _Float16* __restrict__ Vh, int S, int Hn) {
  size_t i = (size_t)blockIdx.x * 256 + threadIdx.x;  // over B*S*H*64
  int d = (int)(i & 63);
  size_t t = i >> 6;
  int h = (int)(t % Hn);
  size_t bs = t / Hn;
  int s = (int)(bs % S);
  int b = (int)(bs / S);
  size_t src = bs * (size_t)(3 * Hn * 64) + (size_t)h * 64 + d;
  size_t dst = (((size_t)b * Hn + h) * S + s) * 64 + d;
  Qh[dst] = qkv[src];
  Kh[dst] = qkv[src + (size_t)Hn * 64];
  Vh[dst] = qkv[src + (size_t)2 * Hn * 64];
}

// x [B*S][H*64] -> dst [B][H][S][64]
__global__ __launch_bounds__(256) void pack_heads_kernel(
    const _Float16* __restrict__ x, _Float16* __restrict__ dst, int S, int Hn) {
  size_t i = (size_t)blockIdx.x * 256 + threadIdx.x;
  int d = (int)(i & 63);
  size_t t = i >> 6;
  int h = (int)(t % Hn);
  size_t bs = t / Hn;
  int s = (int)(bs % S);
  int b = (int)(bs / S);
  dst[(((size_t)b * Hn + h) * S + s) * 64 + d] = x[i];
}

// ---------------------------------------------------------------------------
// Orchestration
// ---------------------------------------------------------------------------
extern "C" void kernel_launch(void* const* d_in, const int* in_sizes, int n_in,
                              void* d_out, int out_size, void* d_ws, size_t ws_size,
                              hipStream_t stream) {
  const int Bc = 2, SQ = 2048, SK = 1024, E = 1024, H = 16, FF = 4096;
  const int HD = H * 64;          // 1024
  const int M1 = Bc * SQ;         // 4096 decoder tokens
  const int M2 = Bc * SK;         // 2048 encoder tokens

  const float* inputs     = (const float*)d_in[0];
  const float* encoded    = (const float*)d_in[1];
  // d_in[2]/d_in[3]: decoder_mask (tril -> causal) / enc-dec mask (all ones);
  // handled structurally by the attention kernels.
  const float* scale_self = (const float*)d_in[4];
  const float* scale_src  = (const float*)d_in[5];
  const float* scale_mlp  = (const float*)d_in[6];
  const float* W_qkv      = (const float*)d_in[7];
  const float* W_self_out = (const float*)d_in[8];
  const float* W_q        = (const float*)d_in[9];
  const float* W_k        = (const float*)d_in[10];
  const float* W_v        = (const float*)d_in[11];
  const float* W_src_out  = (const float*)d_in[12];
  const float* W_fc0      = (const float*)d_in[13];
  const float* W_fc1      = (const float*)d_in[14];
  const float* W_fc_out   = (const float*)d_in[15];
  float* outp = (float*)d_out;

  size_t off = 0;
  auto alloc = [&](size_t bytes) -> void* {
    void* p = (char*)d_ws + off;
    off += (bytes + 255) & ~(size_t)255;
    return p;
  };
  auto h16 = [&](size_t e) { return (_Float16*)alloc(e * 2); };
  auto f32 = [&](size_t e) { return (float*)alloc(e * 4); };

  // --- persistent region (weights f16, residual chain f32, MLP input f16) ---
  _Float16* w_qkv16 = h16((size_t)3 * HD * E);
  _Float16* w_so16  = h16((size_t)E * HD);
  _Float16* w_q16   = h16((size_t)HD * E);
  _Float16* w_k16   = h16((size_t)HD * E);
  _Float16* w_v16   = h16((size_t)HD * E);
  _Float16* w_ro16  = h16((size_t)E * HD);
  _Float16* w_f016  = h16((size_t)FF * E);
  _Float16* w_f116  = h16((size_t)FF * E);
  _Float16* w_fo16  = h16((size_t)E * FF);
  float*    yf32    = f32((size_t)M1 * E);
  _Float16* z16     = h16((size_t)M1 * E);

  // --- phase-A pool (reused for MLP h0/h1/hg afterwards) ---
  size_t pool_off = off;
  _Float16* x16   = h16((size_t)M1 * E);
  _Float16* qkv16 = h16((size_t)M1 * 3 * HD);
  _Float16* Qh    = h16((size_t)M1 * HD);
  _Float16* Kh    = h16((size_t)M1 * HD);
  _Float16* Vh    = h16((size_t)M1 * HD);
  _Float16* a1    = h16((size_t)M1 * HD);
  float*    xf    = f32((size_t)M1 * E);
  _Float16* y16   = h16((size_t)M1 * E);
  _Float16* q2    = h16((size_t)M1 * HD);
  _Float16* Q2p   = h16((size_t)M1 * HD);
  _Float16* enc16 = h16((size_t)M2 * E);
  _Float16* k2    = h16((size_t)M2 * HD);
  _Float16* v2    = h16((size_t)M2 * HD);
  _Float16* K2p   = h16((size_t)M2 * HD);
  _Float16* V2p   = h16((size_t)M2 * HD);
  _Float16* a2    = h16((size_t)M1 * HD);
  // MLP buffers alias the pool (phase-A activations are dead by then;
  // 3 * 32MB = 100.7MB <= pool extent ~138MB).
  size_t hbytes = (((size_t)M1 * FF * 2) + 255) & ~(size_t)255;
  _Float16* h0 = (_Float16*)((char*)d_ws + pool_off);
  _Float16* h1 = (_Float16*)((char*)d_ws + pool_off + hbytes);
  _Float16* hg = (_Float16*)((char*)d_ws + pool_off + 2 * hbytes);
  (void)ws_size; (void)n_in; (void)in_sizes; (void)out_size;

  auto conv = [&](const float* s, _Float16* d, size_t n) {
    f32_to_f16_kernel<<<dim3((unsigned)((n + 1023) / 1024)), 256, 0, stream>>>(s, d, n);
  };
  auto gemm = [&](const _Float16* A, const _Float16* W, const float* resp,
                  float* o32, _Float16* o16, int M, int N, int K) {
    gemm_wmma_kernel<<<dim3(N / 128, M / 256), 256, 0, stream>>>(A, W, resp, o32, o16, M, N, K);
  };

  // Weight / input conversion to f16 (done every call; deterministic).
  conv(W_qkv,      w_qkv16, (size_t)3 * HD * E);
  conv(W_self_out, w_so16,  (size_t)E * HD);
  conv(W_q,        w_q16,   (size_t)HD * E);
  conv(W_k,        w_k16,   (size_t)HD * E);
  conv(W_v,        w_v16,   (size_t)HD * E);
  conv(W_src_out,  w_ro16,  (size_t)E * HD);
  conv(W_fc0,      w_f016,  (size_t)FF * E);
  conv(W_fc1,      w_f116,  (size_t)FF * E);
  conv(W_fc_out,   w_fo16,  (size_t)E * FF);
  conv(encoded,    enc16,   (size_t)M2 * E);

  // ---- self attention block ----
  rmsnorm_f16_kernel<<<M1, 256, 0, stream>>>(inputs, scale_self, x16, E);
  gemm(x16, w_qkv16, nullptr, nullptr, qkv16, M1, 3 * HD, E);
  {
    size_t n = (size_t)M1 * HD;
    pack_qkv_kernel<<<(unsigned)((n + 255) / 256), 256, 0, stream>>>(qkv16, Qh, Kh, Vh, SQ, H);
  }
  attn_wmma_kernel<true><<<dim3(SQ / 64, Bc * H), 128, 0, stream>>>(Qh, Kh, Vh, a1, SQ, SQ, H);
  gemm(a1, w_so16, inputs, xf, nullptr, M1, E, HD);        // x = a@Wso^T + inputs

  // ---- cross attention block ----
  rmsnorm_f16_kernel<<<M1, 256, 0, stream>>>(xf, scale_src, y16, E);
  gemm(y16, w_q16, nullptr, nullptr, q2, M1, HD, E);
  {
    size_t n = (size_t)M1 * HD;
    pack_heads_kernel<<<(unsigned)((n + 255) / 256), 256, 0, stream>>>(q2, Q2p, SQ, H);
  }
  gemm(enc16, w_k16, nullptr, nullptr, k2, M2, HD, E);
  gemm(enc16, w_v16, nullptr, nullptr, v2, M2, HD, E);
  {
    size_t n = (size_t)M2 * HD;
    pack_heads_kernel<<<(unsigned)((n + 255) / 256), 256, 0, stream>>>(k2, K2p, SK, H);
    pack_heads_kernel<<<(unsigned)((n + 255) / 256), 256, 0, stream>>>(v2, V2p, SK, H);
  }
  attn_wmma_kernel<false><<<dim3(SQ / 64, Bc * H), 128, 0, stream>>>(Q2p, K2p, V2p, a2, SQ, SK, H);
  gemm(a2, w_ro16, xf, yf32, nullptr, M1, E, HD);          // y = a@Wro^T + x

  // ---- MLP block (gated exact-GELU) ----
  rmsnorm_f16_kernel<<<M1, 256, 0, stream>>>(yf32, scale_mlp, z16, E);
  gemm(z16, w_f016, nullptr, nullptr, h0, M1, FF, E);
  gemm(z16, w_f116, nullptr, nullptr, h1, M1, FF, E);
  {
    size_t n = (size_t)M1 * FF;
    gelu_gate_kernel<<<(unsigned)((n + 1023) / 1024), 256, 0, stream>>>(h0, h1, hg, n);
  }
  gemm(hg, w_fo16, yf32, outp, nullptr, M1, E, FF);        // z = h@Wfo^T + y
}
